// RWKV_Tmix_x070_58583353917465
// MI455X (gfx1250) — compile-verified, matrix operand order checked by reading, not run on hardware
//
#include <hip/hip_runtime.h>
#include <math.h>

typedef __bf16 bf16;
typedef __attribute__((ext_vector_type(16))) __bf16 v16bf;
typedef __attribute__((ext_vector_type(8)))  __bf16 v8bf;
typedef __attribute__((ext_vector_type(8)))  float  v8f;

#define BB_ 4
#define TT_ 4096
#define CC_ 1024
#define HH_ 16
#define NN_ 64
#define MM_ (BB_ * TT_)          // 16384 rows
#define MC_ ((size_t)MM_ * CC_)  // 16.78M elements
#define DGP_ 192                 // g hidden dim 160 padded to multiple of 64

__device__ __forceinline__ float sigf(float x) { return 1.0f / (1.0f + __expf(-x)); }

// ---------------------------------------------------------------------------
// Weight conversion: f32 -> bf16 (plain), f32 -> bf16 transposed, zero fill
// ---------------------------------------------------------------------------
__global__ void cvt_bf16(const float* __restrict__ in, bf16* __restrict__ out, int n) {
    int i = blockIdx.x * blockDim.x + threadIdx.x;
    if (i < n) out[i] = (bf16)in[i];
}

// in[rows][cols] -> out[cols][rows]  (makes B operand K-contiguous)
__global__ void cvt_bf16_t(const float* __restrict__ in, bf16* __restrict__ out, int rows, int cols) {
    int i = blockIdx.x * blockDim.x + threadIdx.x;
    if (i >= rows * cols) return;
    int r = i / cols, c = i % cols;
    out[(size_t)c * rows + r] = (bf16)in[i];
}

__global__ void fill_zero_bf16(bf16* __restrict__ out, int n) {
    int i = blockIdx.x * blockDim.x + threadIdx.x;
    if (i < n) out[i] = (bf16)0.0f;
}

// ---------------------------------------------------------------------------
// Token shift + 6 lerps -> bf16
// ---------------------------------------------------------------------------
__global__ void mix_kernel(const float* __restrict__ x,
                           const float* __restrict__ mr, const float* __restrict__ mw,
                           const float* __restrict__ mk, const float* __restrict__ mv,
                           const float* __restrict__ ma, const float* __restrict__ mg,
                           bf16* __restrict__ xr, bf16* __restrict__ xw,
                           bf16* __restrict__ xk, bf16* __restrict__ xv,
                           bf16* __restrict__ xa, bf16* __restrict__ xg) {
    size_t i = (size_t)blockIdx.x * blockDim.x + threadIdx.x;
    if (i >= MC_) return;
    int c  = (int)(i % CC_);
    int mt = (int)(i / CC_);
    int t  = mt % TT_;
    float xi = x[i];
    float xp = (t == 0) ? 0.0f : x[i - CC_];
    float xx = xp - xi;
    xr[i] = (bf16)(xi + xx * mr[c]);
    xw[i] = (bf16)(xi + xx * mw[c]);
    xk[i] = (bf16)(xi + xx * mk[c]);
    xv[i] = (bf16)(xi + xx * mv[c]);
    xa[i] = (bf16)(xi + xx * ma[c]);
    xg[i] = (bf16)(xi + xx * mg[c]);
}

// ---------------------------------------------------------------------------
// WMMA bf16 GEMM:  out[M,N] = A[M,K] @ Bn[N,K]^T   (both K-contiguous, NT)
// One wave -> 32(M) x 64(N) tile: 2 A fragments x 4 B fragments = 8 WMMAs
// per k-step (each B fragment reused twice -> 1.5 b128 loads per WMMA).
// N must be a multiple of 64, M of 32; grid is exactly (M/32 * N/64)/8 blocks
// so every wave has a full tile (EXEC stays all-1s through every WMMA).
// ACT: 0 = store f32, 1 = bf16 tanh, 2 = bf16 sigmoid, 3 = bf16 plain
// ---------------------------------------------------------------------------
template <int ACT>
__global__ __launch_bounds__(256)
void gemm_wmma(const bf16* __restrict__ A, const bf16* __restrict__ Bn,
               float* __restrict__ outF, bf16* __restrict__ outB,
               int M, int N, int K, int lda, int ldb, int ldc) {
    const int tilesN = N >> 6;
    // wave-uniform tile index, forced scalar
    const int wid    = __builtin_amdgcn_readfirstlane((int)(threadIdx.x >> 5));
    const int tileId = blockIdx.x * 8 + wid;
    const int tm = tileId / tilesN, tn = tileId % tilesN;
    const int row0 = tm << 5, n0 = tn << 6;

    const int lane = threadIdx.x & 31;
    const int lrow = lane & 15;
    const int lhi  = lane >> 4;
    const int kb   = lhi << 3;   // 0 or 8

    const bf16* A0 = A + (size_t)(row0 +  0 + lrow) * lda;
    const bf16* A1 = A + (size_t)(row0 + 16 + lrow) * lda;
    const bf16* B0 = Bn + (size_t)(n0 +  0 + lrow) * ldb;
    const bf16* B1 = Bn + (size_t)(n0 + 16 + lrow) * ldb;
    const bf16* B2 = Bn + (size_t)(n0 + 32 + lrow) * ldb;
    const bf16* B3 = Bn + (size_t)(n0 + 48 + lrow) * ldb;

    v8f zero = {0.f, 0.f, 0.f, 0.f, 0.f, 0.f, 0.f, 0.f};
    v8f acc00 = zero, acc01 = zero, acc02 = zero, acc03 = zero;
    v8f acc10 = zero, acc11 = zero, acc12 = zero, acc13 = zero;

    union Frag { v16bf v; v8bf h[2]; };

    for (int k0 = 0; k0 < K; k0 += 32) {
        Frag a0, a1, b0, b1, b2, b3;
        a0.h[0] = *(const v8bf*)(A0 + k0 + kb);
        a0.h[1] = *(const v8bf*)(A0 + k0 + 16 + kb);
        a1.h[0] = *(const v8bf*)(A1 + k0 + kb);
        a1.h[1] = *(const v8bf*)(A1 + k0 + 16 + kb);
        b0.h[0] = *(const v8bf*)(B0 + k0 + kb);
        b0.h[1] = *(const v8bf*)(B0 + k0 + 16 + kb);
        b1.h[0] = *(const v8bf*)(B1 + k0 + kb);
        b1.h[1] = *(const v8bf*)(B1 + k0 + 16 + kb);
        b2.h[0] = *(const v8bf*)(B2 + k0 + kb);
        b2.h[1] = *(const v8bf*)(B2 + k0 + 16 + kb);
        b3.h[0] = *(const v8bf*)(B3 + k0 + kb);
        b3.h[1] = *(const v8bf*)(B3 + k0 + 16 + kb);
        acc00 = __builtin_amdgcn_wmma_f32_16x16x32_bf16(false, a0.v, false, b0.v, (short)0, acc00, false, false);
        acc10 = __builtin_amdgcn_wmma_f32_16x16x32_bf16(false, a1.v, false, b0.v, (short)0, acc10, false, false);
        acc01 = __builtin_amdgcn_wmma_f32_16x16x32_bf16(false, a0.v, false, b1.v, (short)0, acc01, false, false);
        acc11 = __builtin_amdgcn_wmma_f32_16x16x32_bf16(false, a1.v, false, b1.v, (short)0, acc11, false, false);
        acc02 = __builtin_amdgcn_wmma_f32_16x16x32_bf16(false, a0.v, false, b2.v, (short)0, acc02, false, false);
        acc12 = __builtin_amdgcn_wmma_f32_16x16x32_bf16(false, a1.v, false, b2.v, (short)0, acc12, false, false);
        acc03 = __builtin_amdgcn_wmma_f32_16x16x32_bf16(false, a0.v, false, b3.v, (short)0, acc03, false, false);
        acc13 = __builtin_amdgcn_wmma_f32_16x16x32_bf16(false, a1.v, false, b3.v, (short)0, acc13, false, false);
    }

    v8f accs[2][4] = {{acc00, acc01, acc02, acc03}, {acc10, acc11, acc12, acc13}};
#pragma unroll
    for (int r = 0; r < 2; ++r) {
#pragma unroll
        for (int s = 0; s < 4; ++s) {
            int col = n0 + s * 16 + lrow;
#pragma unroll
            for (int v = 0; v < 8; ++v) {
                int row = row0 + r * 16 + v + (lhi << 3);
                size_t o = (size_t)row * ldc + col;
                float x = accs[r][s][v];
                if (ACT == 0)      outF[o] = x;
                else if (ACT == 1) outB[o] = (bf16)tanhf(x);
                else if (ACT == 2) outB[o] = (bf16)sigf(x);
                else               outB[o] = (bf16)x;
            }
        }
    }
}

// ---------------------------------------------------------------------------
// Elementwise epilogues
// ---------------------------------------------------------------------------
__global__ void ew_decay(float* __restrict__ Wd, const float* __restrict__ w0) {
    size_t i = (size_t)blockIdx.x * blockDim.x + threadIdx.x;
    if (i >= MC_) return;
    int c = (int)(i % CC_);
    // w = -softplus(-(w0+pre)) - 0.5 ;  decay = exp(-exp(w)) = exp(-e^{-1/2} * sigmoid(u))
    float u = w0[c] + Wd[i];
    Wd[i] = __expf(-0.60653066f * sigf(u));
}

__global__ void ew_asig(float* __restrict__ Ab, const float* __restrict__ a0) {
    size_t i = (size_t)blockIdx.x * blockDim.x + threadIdx.x;
    if (i >= MC_) return;
    int c = (int)(i % CC_);
    Ab[i] = sigf(a0[c] + Ab[i]);
}

__global__ void ew_vlerp(float* __restrict__ Vb, const float* __restrict__ VP,
                         const float* __restrict__ v_first, const float* __restrict__ v0) {
    size_t i = (size_t)blockIdx.x * blockDim.x + threadIdx.x;
    if (i >= MC_) return;
    int c = (int)(i % CC_);
    float s = sigf(v0[c] + VP[i]);
    float v = Vb[i];
    Vb[i] = v + (v_first[i] - v) * s;
}

// ---------------------------------------------------------------------------
// Per-head kk normalize + k rescale.  One 64-thread block per (token, head).
// Writes NK = -kk_norm, Bm = kk_norm * a, Kb <- k*(1+(a-1)*k_a)  (in place)
// ---------------------------------------------------------------------------
__global__ void kk_kernel(float* __restrict__ Kb, const float* __restrict__ Ab,
                          const float* __restrict__ k_k, const float* __restrict__ k_a,
                          float* __restrict__ NK, float* __restrict__ Bm) {
    int row = blockIdx.x;            // [0, M*H)
    int h   = row % HH_;
    int mt  = row / HH_;
    int j   = threadIdx.x;
    size_t base = (size_t)mt * CC_ + h * NN_;
    int c = h * NN_ + j;

    float kraw = Kb[base + j];
    float a    = Ab[base + j];
    float kk   = kraw * k_k[c];

    __shared__ float s[NN_];
    s[j] = kk * kk;
    __syncthreads();
    for (int off = 32; off; off >>= 1) {
        if (j < off) s[j] += s[j + off];
        __syncthreads();
    }
    float inv = 1.0f / fmaxf(sqrtf(s[0]), 1e-12f);
    float kkn = kk * inv;
    NK[base + j] = -kkn;
    Bm[base + j] = kkn * a;
    Kb[base + j] = kraw * (1.0f + (a - 1.0f) * k_a[c]);
}

// ---------------------------------------------------------------------------
// RWKV7 recurrence: one 64-thread block per (b,h); thread i owns state row i
// (64 f32 in registers). Serial over T, vectors staged through LDS.
// Next step's vectors are prefetched into registers during the current step's
// compute to hide HBM latency; dot-product chains split 4-way.
// ---------------------------------------------------------------------------
__global__ __launch_bounds__(64)
void rwkv_recurrence(const float* __restrict__ R,  const float* __restrict__ Wd,
                     const float* __restrict__ Kf, const float* __restrict__ Vf,
                     const float* __restrict__ NK, const float* __restrict__ Bm,
                     float* __restrict__ Y) {
    const int b = blockIdx.x / HH_;
    const int h = blockIdx.x % HH_;
    const int i = threadIdx.x;

    __shared__ float shr[NN_], shw[NN_], shk[NN_], sha[NN_], shb[NN_];

    float st[NN_];
#pragma unroll
    for (int j = 0; j < NN_; ++j) st[j] = 0.0f;

    const size_t base0 = ((size_t)b * TT_) * CC_ + h * NN_ + i;

    // prefetch t = 0
    float pr = R[base0],  pw = Wd[base0], pk = Kf[base0];
    float pa = NK[base0], pb = Bm[base0], pv = Vf[base0];

    for (int t = 0; t < TT_; ++t) {
        shr[i] = pr; shw[i] = pw; shk[i] = pk; sha[i] = pa; shb[i] = pb;
        float v = pv;
        __syncthreads();

        // issue next step's loads; latency overlaps the FMA work below
        if (t + 1 < TT_) {
            size_t nb = base0 + (size_t)(t + 1) * CC_;
            pr = R[nb];  pw = Wd[nb]; pk = Kf[nb];
            pa = NK[nb]; pb = Bm[nb]; pv = Vf[nb];
        }

        float sa0 = 0.f, sa1 = 0.f, sa2 = 0.f, sa3 = 0.f;
#pragma unroll
        for (int j = 0; j < NN_; j += 4) {
            sa0 += st[j]     * sha[j];
            sa1 += st[j + 1] * sha[j + 1];
            sa2 += st[j + 2] * sha[j + 2];
            sa3 += st[j + 3] * sha[j + 3];
        }
        float sa = (sa0 + sa1) + (sa2 + sa3);

        float o0 = 0.f, o1 = 0.f, o2 = 0.f, o3 = 0.f;
#pragma unroll
        for (int j = 0; j < NN_; j += 4) {
            st[j]     = st[j]     * shw[j]     + sa * shb[j]     + v * shk[j];     o0 += st[j]     * shr[j];
            st[j + 1] = st[j + 1] * shw[j + 1] + sa * shb[j + 1] + v * shk[j + 1]; o1 += st[j + 1] * shr[j + 1];
            st[j + 2] = st[j + 2] * shw[j + 2] + sa * shb[j + 2] + v * shk[j + 2]; o2 += st[j + 2] * shr[j + 2];
            st[j + 3] = st[j + 3] * shw[j + 3] + sa * shb[j + 3] + v * shk[j + 3]; o3 += st[j + 3] * shr[j + 3];
        }
        Y[base0 + (size_t)t * CC_] = (o0 + o1) + (o2 + o3);
        __syncthreads();
    }
}

// ---------------------------------------------------------------------------
// GroupNorm(head) + rk*v residual + gate -> bf16
// ---------------------------------------------------------------------------
__global__ void gn_kernel(const float* __restrict__ Y,  const float* __restrict__ R,
                          const float* __restrict__ Kf, const float* __restrict__ Vf,
                          const float* __restrict__ G,
                          const float* __restrict__ ln_w, const float* __restrict__ ln_b,
                          const float* __restrict__ r_k,  bf16* __restrict__ out) {
    int row = blockIdx.x;            // [0, M*H)
    int h   = row % HH_;
    int mt  = row / HH_;
    int j   = threadIdx.x;
    size_t base = (size_t)mt * CC_ + h * NN_;
    int c = h * NN_ + j;

    float y  = Y[base + j];
    float rk = R[base + j] * Kf[base + j] * r_k[c];

    __shared__ float s0[NN_], s1[NN_], s2[NN_];
    s0[j] = y; s1[j] = y * y; s2[j] = rk;
    __syncthreads();
    for (int off = 32; off; off >>= 1) {
        if (j < off) { s0[j] += s0[j + off]; s1[j] += s1[j + off]; s2[j] += s2[j + off]; }
        __syncthreads();
    }
    float mu  = s0[0] * (1.0f / NN_);
    float var = s1[0] * (1.0f / NN_) - mu * mu;
    float yn  = (y - mu) * rsqrtf(var + 0.00064f) * ln_w[c] + ln_b[c];
    yn += s2[0] * Vf[base + j];
    out[base + j] = (bf16)(yn * G[base + j]);
}

// ---------------------------------------------------------------------------
// Host orchestration
// ---------------------------------------------------------------------------
extern "C" void kernel_launch(void* const* d_in, const int* in_sizes, int n_in,
                              void* d_out, int out_size, void* d_ws, size_t ws_size,
                              hipStream_t stream) {
    const float* x       = (const float*)d_in[0];
    const float* v_first = (const float*)d_in[1];
    const float* x_r     = (const float*)d_in[2];
    const float* x_w     = (const float*)d_in[3];
    const float* x_k     = (const float*)d_in[4];
    const float* x_v     = (const float*)d_in[5];
    const float* x_a     = (const float*)d_in[6];
    const float* x_g     = (const float*)d_in[7];
    const float* w0      = (const float*)d_in[8];
    const float* w1      = (const float*)d_in[9];
    const float* w2      = (const float*)d_in[10];
    const float* a0      = (const float*)d_in[11];
    const float* a1      = (const float*)d_in[12];
    const float* a2      = (const float*)d_in[13];
    const float* v0      = (const float*)d_in[14];
    const float* v1      = (const float*)d_in[15];
    const float* v2      = (const float*)d_in[16];
    const float* g1      = (const float*)d_in[17];
    const float* g2      = (const float*)d_in[18];
    const float* k_k     = (const float*)d_in[19];
    const float* k_a     = (const float*)d_in[20];
    const float* r_k     = (const float*)d_in[21];
    const float* W_r     = (const float*)d_in[22];
    const float* W_k     = (const float*)d_in[23];
    const float* W_v     = (const float*)d_in[24];
    const float* W_o     = (const float*)d_in[25];
    const float* ln_w    = (const float*)d_in[26];
    const float* ln_b    = (const float*)d_in[27];
    float* out = (float*)d_out;

    char* wsp = (char*)d_ws;
    size_t off = 0;
    auto alloc = [&](size_t bytes) -> void* {
        void* p = wsp + off;
        off = (off + bytes + 255) & ~(size_t)255;
        return p;
    };

    const int D = 64, DG = 160, C = CC_, M = MM_;

    // bf16 activation buffers
    bf16* xr = (bf16*)alloc(MC_ * 2); bf16* xw = (bf16*)alloc(MC_ * 2);
    bf16* xk = (bf16*)alloc(MC_ * 2); bf16* xv = (bf16*)alloc(MC_ * 2);
    bf16* xa = (bf16*)alloc(MC_ * 2); bf16* xg = (bf16*)alloc(MC_ * 2);
    bf16* yng = (bf16*)alloc(MC_ * 2);
    // bf16 weights (K-contiguous)
    bf16* bWr = (bf16*)alloc((size_t)C * C * 2); bf16* bWk = (bf16*)alloc((size_t)C * C * 2);
    bf16* bWv = (bf16*)alloc((size_t)C * C * 2); bf16* bWo = (bf16*)alloc((size_t)C * C * 2);
    bf16* w1t = (bf16*)alloc((size_t)D * C * 2);  bf16* w2t = (bf16*)alloc((size_t)C * D * 2);
    bf16* a1t = (bf16*)alloc((size_t)D * C * 2);  bf16* a2t = (bf16*)alloc((size_t)C * D * 2);
    bf16* v1t = (bf16*)alloc((size_t)D * C * 2);  bf16* v2t = (bf16*)alloc((size_t)C * D * 2);
    bf16* g1t = (bf16*)alloc((size_t)DGP_ * C * 2);   // 192 rows, rows 160..191 zero
    bf16* g2t = (bf16*)alloc((size_t)C * DG * 2);
    // bf16 LoRA hiddens
    bf16* hw = (bf16*)alloc((size_t)M * D * 2);
    bf16* ha = (bf16*)alloc((size_t)M * D * 2);
    bf16* hv = (bf16*)alloc((size_t)M * D * 2);
    bf16* hg = (bf16*)alloc((size_t)M * DGP_ * 2);    // padded to 192 cols
    // f32 full-size buffers
    float* R  = (float*)alloc(MC_ * 4); float* Wd = (float*)alloc(MC_ * 4);
    float* Kb = (float*)alloc(MC_ * 4); float* Vb = (float*)alloc(MC_ * 4);
    float* Ab = (float*)alloc(MC_ * 4); float* VP = (float*)alloc(MC_ * 4);
    float* G  = (float*)alloc(MC_ * 4); float* NK = (float*)alloc(MC_ * 4);
    float* Bm = (float*)alloc(MC_ * 4); float* Y  = (float*)alloc(MC_ * 4);
    (void)ws_size; (void)in_sizes; (void)n_in; (void)out_size;

    auto eblk = [](size_t n) { return (int)((n + 255) / 256); };
    auto gblk = [](int Mm, int Nn) {        // M mult of 32, N mult of 64; exact
        return (Mm / 32) * (Nn / 64) / 8;
    };

    // --- weight prep ---
    cvt_bf16<<<eblk((size_t)C * C), 256, 0, stream>>>(W_r, bWr, C * C);
    cvt_bf16<<<eblk((size_t)C * C), 256, 0, stream>>>(W_k, bWk, C * C);
    cvt_bf16<<<eblk((size_t)C * C), 256, 0, stream>>>(W_v, bWv, C * C);
    cvt_bf16<<<eblk((size_t)C * C), 256, 0, stream>>>(W_o, bWo, C * C);
    cvt_bf16_t<<<eblk((size_t)C * D), 256, 0, stream>>>(w1, w1t, C, D);
    cvt_bf16_t<<<eblk((size_t)C * D), 256, 0, stream>>>(w2, w2t, D, C);
    cvt_bf16_t<<<eblk((size_t)C * D), 256, 0, stream>>>(a1, a1t, C, D);
    cvt_bf16_t<<<eblk((size_t)C * D), 256, 0, stream>>>(a2, a2t, D, C);
    cvt_bf16_t<<<eblk((size_t)C * D), 256, 0, stream>>>(v1, v1t, C, D);
    cvt_bf16_t<<<eblk((size_t)C * D), 256, 0, stream>>>(v2, v2t, D, C);
    fill_zero_bf16<<<eblk((size_t)DGP_ * C), 256, 0, stream>>>(g1t, DGP_ * C);
    cvt_bf16_t<<<eblk((size_t)C * DG), 256, 0, stream>>>(g1, g1t, C, DG);
    cvt_bf16_t<<<eblk((size_t)C * DG), 256, 0, stream>>>(g2, g2t, DG, C);

    // --- token shift + lerps ---
    mix_kernel<<<eblk(MC_), 256, 0, stream>>>(x, x_r, x_w, x_k, x_v, x_a, x_g,
                                              xr, xw, xk, xv, xa, xg);

    // --- big projections (WMMA) ---
    gemm_wmma<0><<<gblk(M, C), 256, 0, stream>>>(xr, bWr, R,  nullptr, M, C, C, C, C, C);
    gemm_wmma<0><<<gblk(M, C), 256, 0, stream>>>(xk, bWk, Kb, nullptr, M, C, C, C, C, C);
    gemm_wmma<0><<<gblk(M, C), 256, 0, stream>>>(xv, bWv, Vb, nullptr, M, C, C, C, C, C);

    // --- w path: tanh(xw@w1) @ w2, then decay ---
    gemm_wmma<1><<<gblk(M, D), 256, 0, stream>>>(xw, w1t, nullptr, hw, M, D, C, C, C, D);
    gemm_wmma<0><<<gblk(M, C), 256, 0, stream>>>(hw, w2t, Wd, nullptr, M, C, D, D, D, C);
    ew_decay<<<eblk(MC_), 256, 0, stream>>>(Wd, w0);

    // --- a path: sigmoid(a0 + (xa@a1)@a2) ---
    gemm_wmma<3><<<gblk(M, D), 256, 0, stream>>>(xa, a1t, nullptr, ha, M, D, C, C, C, D);
    gemm_wmma<0><<<gblk(M, C), 256, 0, stream>>>(ha, a2t, Ab, nullptr, M, C, D, D, D, C);
    ew_asig<<<eblk(MC_), 256, 0, stream>>>(Ab, a0);

    // --- v path: lerp toward v_first ---
    gemm_wmma<3><<<gblk(M, D), 256, 0, stream>>>(xv, v1t, nullptr, hv, M, D, C, C, C, D);
    gemm_wmma<0><<<gblk(M, C), 256, 0, stream>>>(hv, v2t, VP, nullptr, M, C, D, D, D, C);
    ew_vlerp<<<eblk(MC_), 256, 0, stream>>>(Vb, VP, v_first, v0);

    // --- g path: sigmoid(xg@g1) @ g2  (hidden padded 160 -> 192) ---
    gemm_wmma<2><<<gblk(M, DGP_), 256, 0, stream>>>(xg, g1t, nullptr, hg, M, DGP_, C, C, C, DGP_);
    gemm_wmma<0><<<gblk(M, C),    256, 0, stream>>>(hg, g2t, G, nullptr, M, C, DG, DGP_, DG, C);

    // --- kk normalize, k rescale, a/b vectors ---
    kk_kernel<<<M * HH_, NN_, 0, stream>>>(Kb, Ab, k_k, k_a, NK, Bm);

    // --- serial recurrence ---
    rwkv_recurrence<<<BB_ * HH_, NN_, 0, stream>>>(R, Wd, Kb, Vb, NK, Bm, Y);

    // --- GroupNorm + rk*v residual + gate ---
    gn_kernel<<<M * HH_, NN_, 0, stream>>>(Y, R, Kb, Vb, G, ln_w, ln_b, r_k, yng);

    // --- output projection (WMMA) -> f32 out ---
    gemm_wmma<0><<<gblk(M, C), 256, 0, stream>>>(yng, bWo, out, nullptr, M, C, C, C, C, C);
}
